// GAT_56710748176501
// MI455X (gfx1250) — matrix operand order repaired and checked by reading, not compiled, not run on hardware
//
#include <hip/hip_runtime.h>

// ---------------------------------------------------------------------------
// Shapes (match reference)
// ---------------------------------------------------------------------------
constexpr int NB    = 64;          // graphs
constexpr int NODES = 512;         // nodes per graph
constexpr int NN    = NB * NODES;  // 32768 total nodes
constexpr int INF_  = 128;         // input feature dim
constexpr int HH    = 4;           // heads
constexpr int DD    = 50;          // feats per head
constexpr int HD    = 200;         // H*D
constexpr int LDH   = 208;         // padded GEMM N (13 * 16)
constexpr int KPAD  = 224;         // padded GEMM K for 200 (7 * 32)
constexpr int NTIL  = LDH / 16;    // 13 column tiles
constexpr int NTPAD = 16;          // padded n-tiles in fragment-packed B
constexpr int P1L   = 254;         // pooled length after conv1+pool(3,2)
constexpr int P1R   = NB * P1L;    // 16256 rows (1016 * 16)
constexpr float NEG_SLOPE = 0.2f;

typedef _Float16 h8   __attribute__((ext_vector_type(8)));
typedef _Float16 v16h __attribute__((ext_vector_type(16)));
typedef float    v8f  __attribute__((ext_vector_type(8)));

// ---------------------------------------------------------------------------
// WMMA GEMM: C[M,ldc](f32) (+)= A[M,lda](f16) @ B(f16, fragment-packed)
// One wave computes a 16-row stripe x NACT column tiles. NACT is a template
// parameter so the K-loop is branch-free: 1 A-fragment load, NACT contiguous
// B-fragment loads (32B v16h per lane, coalesced), NACT back-to-back WMMAs.
// A/B/C VGPR layouts per CDNA5 ISA §7.12.2.
// ---------------------------------------------------------------------------
template <int NACT>
__device__ __forceinline__ void gemm_stripe(const _Float16* __restrict__ A, int lda,
                                            const _Float16* __restrict__ Bp,
                                            float* __restrict__ C, int ldc,
                                            int tile_m, int nt0, int K,
                                            int accumulate, int lane) {
  const int mrow  = tile_m + (lane & 15);
  const int hiA   = (lane >> 4) * 8;          // A K-offset by lane half
  const int rbase = tile_m + (lane >> 4) * 8;
  const int ncol0 = (lane & 15);

  v8f acc[NACT];
#pragma unroll
  for (int j = 0; j < NACT; ++j) acc[j] = {};
  if (accumulate) {
#pragma unroll
    for (int j = 0; j < NACT; ++j) {
      const int cn = (nt0 + j) * 16 + ncol0;
#pragma unroll
      for (int r = 0; r < 8; ++r)
        acc[j][r] = C[(size_t)(rbase + r) * ldc + cn];
    }
  }

  const int kTiles = K >> 5;
  for (int kt = 0; kt < kTiles; ++kt) {
    const _Float16* arow = A + (size_t)mrow * lda + kt * 32 + hiA;
    h8 alo = *(const h8*)(arow);              // K = kt*32+hiA+0..7   -> elems 0..7
    h8 ahi = *(const h8*)(arow + 16);         // K = kt*32+hiA+16..23 -> elems 8..15
    v16h af;
#pragma unroll
    for (int i = 0; i < 8; ++i) { af[i] = alo[i]; af[i + 8] = ahi[i]; }

    const _Float16* bbase = Bp + (((size_t)kt * NTPAD + nt0) * 32 + lane) * 16;
#pragma unroll
    for (int j = 0; j < NACT; ++j) {
      v16h bf = *(const v16h*)(bbase + j * 512);   // n-tile stride: 32 lanes * 16h
      acc[j] = __builtin_amdgcn_wmma_f32_16x16x32_f16(false, af, false, bf,
                                                      (short)0, acc[j], false, false);
    }
  }

#pragma unroll
  for (int j = 0; j < NACT; ++j) {
    const int cn = (nt0 + j) * 16 + ncol0;
#pragma unroll
    for (int r = 0; r < 8; ++r)
      C[(size_t)(rbase + r) * ldc + cn] = acc[j][r];
  }
}

__global__ void k_wmma_gemm(const _Float16* __restrict__ A, int lda,
                            const _Float16* __restrict__ Bp,
                            float* __restrict__ C, int ldc,
                            int nTiles, int K, int accumulate) {
  const int lane = threadIdx.x;               // 0..31 (wave32)
  const int nt0  = threadIdx.y * 4;           // blockDim.y==4 covers 16 n-tiles
  const int nAct = nTiles - nt0;
  if (nAct <= 0) return;
  const int tile_m = blockIdx.x * 16;
  if (nAct >= 4)
    gemm_stripe<4>(A, lda, Bp, C, ldc, tile_m, nt0, K, accumulate, lane);
  else if (nAct == 3)
    gemm_stripe<3>(A, lda, Bp, C, ldc, tile_m, nt0, K, accumulate, lane);
  else if (nAct == 2)
    gemm_stripe<2>(A, lda, Bp, C, ldc, tile_m, nt0, K, accumulate, lane);
  else
    gemm_stripe<1>(A, lda, Bp, C, ldc, tile_m, nt0, K, accumulate, lane);
}

// ---------------------------------------------------------------------------
// Packing kernels (f32 -> zero-padded f16)
// ---------------------------------------------------------------------------
__global__ void k_pack_a(const float* __restrict__ src, _Float16* __restrict__ dst,
                         int srcRows, int srcCols, int dstCols, long total) {
  long idx = (long)blockIdx.x * blockDim.x + threadIdx.x;
  if (idx >= total) return;
  int r = (int)(idx / dstCols), c = (int)(idx % dstCols);
  float v = (r < srcRows && c < srcCols) ? src[(size_t)r * srcCols + c] : 0.f;
  dst[idx] = (_Float16)v;
}

// Fragment-packed weights: dst[((kt*NTPAD + nt)*32 + lane)*16 + i] holds
// B(k = kt*32 + (lane>>4)*8 + (i&7) + (i>>3)*16, n = nt*16 + (lane&15)), B(k,n)=src[k*sr+n*sc+off]
__global__ void k_pack_w_frag(const float* __restrict__ src, _Float16* __restrict__ dst,
                              int rows, int cols, int sr, int sc, int off, int kTiles) {
  long idx = (long)blockIdx.x * blockDim.x + threadIdx.x;
  long total = (long)kTiles * NTPAD * 32 * 16;
  if (idx >= total) return;
  int i    = (int)(idx & 15);
  int lane = (int)((idx >> 4) & 31);
  int nt   = (int)((idx >> 9) & 15);
  int kt   = (int)(idx >> 13);
  int k = kt * 32 + ((lane >> 4) << 3) + (i & 7) + ((i >> 3) << 4);
  int n = nt * 16 + (lane & 15);
  float v = (k < rows && n < cols) ? src[(size_t)k * sr + (size_t)n * sc + off] : 0.f;
  dst[idx] = (_Float16)v;
}

__global__ void k_zero(float* p, long n) {
  long idx = (long)blockIdx.x * blockDim.x + threadIdx.x;
  if (idx < n) p[idx] = 0.f;
}

// ---------------------------------------------------------------------------
// GAT attention kernels
// ---------------------------------------------------------------------------
__device__ __forceinline__ unsigned encf(float f) {
  unsigned u = __float_as_uint(f);
  return (u & 0x80000000u) ? ~u : (u | 0x80000000u);
}
__device__ __forceinline__ float decf(unsigned u) {
  return __uint_as_float((u & 0x80000000u) ? (u ^ 0x80000000u) : ~u);
}

__global__ void k_scores(const float* __restrict__ hbuf,
                         const float* __restrict__ al, const float* __restrict__ ar,
                         float* __restrict__ el, float* __restrict__ er,
                         unsigned* __restrict__ mEnc, float* __restrict__ z) {
  int idx = blockIdx.x * blockDim.x + threadIdx.x;  // NN*H
  if (idx >= NN * HH) return;
  int n = idx >> 2, hh = idx & 3;
  const float* hp  = hbuf + (size_t)n * LDH + hh * DD;
  const float* alp = al + hh * DD;
  const float* arp = ar + hh * DD;
  float sl = 0.f, sr = 0.f;
#pragma unroll 10
  for (int j = 0; j < DD; ++j) { sl += hp[j] * alp[j]; sr += hp[j] * arp[j]; }
  el[idx] = sl; er[idx] = sr;
  mEnc[idx] = encf(-__builtin_inff());
  z[idx] = 0.f;
}

__device__ __forceinline__ void edge_sd(int e, int E, const int* src, const int* dst,
                                        int& s_, int& d_) {
  if (e < E) { s_ = src[e]; d_ = dst[e]; }
  else       { s_ = e - E;  d_ = e - E; }   // self loops
}

__global__ void k_edge_max(const int* __restrict__ src, const int* __restrict__ dst, int E,
                           const float* __restrict__ el, const float* __restrict__ er,
                           unsigned* __restrict__ mEnc, long total) {
  long idx = (long)blockIdx.x * blockDim.x + threadIdx.x;
  if (idx >= total) return;
  int e = (int)(idx >> 2), hh = (int)(idx & 3);
  int s_, d_; edge_sd(e, E, src, dst, s_, d_);
  float ev = el[s_ * HH + hh] + er[d_ * HH + hh];
  ev = ev > 0.f ? ev : NEG_SLOPE * ev;
  atomicMax(&mEnc[d_ * HH + hh], encf(ev));
}

__global__ void k_edge_expsum(const int* __restrict__ src, const int* __restrict__ dst, int E,
                              const float* __restrict__ el, const float* __restrict__ er,
                              const unsigned* __restrict__ mEnc,
                              float* __restrict__ exBuf, float* __restrict__ z, long total) {
  long idx = (long)blockIdx.x * blockDim.x + threadIdx.x;
  if (idx >= total) return;
  int e = (int)(idx >> 2), hh = (int)(idx & 3);
  int s_, d_; edge_sd(e, E, src, dst, s_, d_);
  float ev = el[s_ * HH + hh] + er[d_ * HH + hh];
  ev = ev > 0.f ? ev : NEG_SLOPE * ev;
  float ex = __expf(ev - decf(mEnc[d_ * HH + hh]));
  exBuf[idx] = ex;
  atomicAdd(&z[d_ * HH + hh], ex);
}

__global__ void k_edge_agg(const int* __restrict__ src, const int* __restrict__ dst, int E,
                           const float* __restrict__ exBuf, const float* __restrict__ z,
                           const float* __restrict__ hbuf, float* __restrict__ agg,
                           long total) {
  long idx = (long)blockIdx.x * blockDim.x + threadIdx.x;
  if (idx >= total) return;
  int e = (int)(idx >> 2), hh = (int)(idx & 3);
  int s_, d_; edge_sd(e, E, src, dst, s_, d_);
  float alpha = exBuf[idx] / z[d_ * HH + hh];
  const float* hp = hbuf + (size_t)s_ * LDH + hh * DD;
  float* op = agg + (size_t)d_ * HD + hh * DD;
#pragma unroll 10
  for (int j = 0; j < DD; ++j) atomicAdd(&op[j], hp[j] * alpha);
}

__global__ void k_bias_relu_acc(const float* __restrict__ agg, const float* __restrict__ b,
                                float* __restrict__ fcur, float* __restrict__ f5) {
  long idx = (long)blockIdx.x * blockDim.x + threadIdx.x;
  if (idx >= (long)NN * HD) return;
  int c = (int)(idx % HD);
  float v = fmaxf(agg[idx] + b[c], 0.f);
  fcur[idx] = v;
  f5[idx] += v;
}

// ---------------------------------------------------------------------------
// Conv head kernels
// ---------------------------------------------------------------------------
// relu(conv1 + bias) then maxpool(3,2); p1 laid out [b*254+lp, 200]
__global__ void k_conv_pool1(const float* __restrict__ convo, const float* __restrict__ cb1,
                             float* __restrict__ p1) {
  long idx = (long)blockIdx.x * blockDim.x + threadIdx.x;
  if (idx >= (long)P1R * HD) return;
  int co = (int)(idx % HD);
  int t  = (int)(idx / HD);           // b*254 + lp
  int lp = t % P1L, b = t / P1L;
  float bias = cb1[co];
  size_t base = ((size_t)b * NODES + 2 * lp) * LDH + co;
  float mv = 0.f;                      // relu outputs are >= 0
#pragma unroll
  for (int j = 0; j < 3; ++j) mv = fmaxf(mv, convo[base + (size_t)j * LDH] + bias);
  p1[(size_t)t * HD + co] = mv;
}

// relu(conv2 + bias), maxpool(2,2) over 254 -> 127, sum over positions
__global__ void k_pool2_sum(const float* __restrict__ c2, const float* __restrict__ cb2,
                            float* __restrict__ ssum) {
  int idx = blockIdx.x * blockDim.x + threadIdx.x;   // NB*HD
  if (idx >= NB * HD) return;
  int co = idx % HD, b = idx / HD;
  float bias = cb2[co], s = 0.f;
  for (int lp = 0; lp < 127; ++lp) {
    size_t base = ((size_t)b * P1L + 2 * lp) * LDH + co;
    float v0 = fmaxf(c2[base] + bias, 0.f);
    float v1 = fmaxf(c2[base + LDH] + bias, 0.f);
    s += fmaxf(v0, v1);
  }
  ssum[idx] = s;
}

__global__ void k_classifier(const float* __restrict__ ssum, const float* __restrict__ lw,
                             const float* __restrict__ lb, float* __restrict__ out) {
  int b = blockIdx.x * blockDim.x + threadIdx.x;
  if (b >= NB) return;
  float acc = lb[0];
  for (int c = 0; c < HD; ++c) acc += ssum[b * HD + c] * lw[c];
  out[b] = 1.f / (1.f + __expf(-acc));
}

// ---------------------------------------------------------------------------
// Host orchestration
// ---------------------------------------------------------------------------
static inline dim3 g1(long n, int blk = 256) { return dim3((unsigned)((n + blk - 1) / blk)); }

static void gemm(const _Float16* A, int lda, const _Float16* Bp,
                 float* C, int ldc, int M, int nTiles, int K, int accum,
                 hipStream_t s) {
  dim3 blk(32, 4);                 // one block covers up to 16 n-tiles
  dim3 grd(M / 16, 1);
  k_wmma_gemm<<<grd, blk, 0, s>>>(A, lda, Bp, C, ldc, nTiles, K, accum);
}

extern "C" void kernel_launch(void* const* d_in, const int* in_sizes, int n_in,
                              void* d_out, int out_size, void* d_ws, size_t ws_size,
                              hipStream_t stream) {
  const float* x   = (const float*)d_in[0];
  const int*   src = (const int*)d_in[1];
  const int*   dst = (const int*)d_in[2];
  const int    E   = in_sizes[1];
  const long   ET  = (long)(E + NN) * HH;

  const float* Wl[4]  = {(const float*)d_in[3], (const float*)d_in[7],
                         (const float*)d_in[11], (const float*)d_in[11]};
  const float* all[4] = {(const float*)d_in[4], (const float*)d_in[8],
                         (const float*)d_in[12], (const float*)d_in[12]};
  const float* arl[4] = {(const float*)d_in[5], (const float*)d_in[9],
                         (const float*)d_in[13], (const float*)d_in[13]};
  const float* bl[4]  = {(const float*)d_in[6], (const float*)d_in[10],
                         (const float*)d_in[14], (const float*)d_in[14]};
  const float* cw1 = (const float*)d_in[15];
  const float* cb1 = (const float*)d_in[16];
  const float* cw2 = (const float*)d_in[17];
  const float* cb2 = (const float*)d_in[18];
  const float* lw  = (const float*)d_in[19];
  const float* lb  = (const float*)d_in[20];
  float* out = (float*)d_out;

  // bump allocator over workspace
  char* ws = (char*)d_ws;
  auto alloc = [&](size_t bytes) -> void* {
    void* p = ws; ws += (bytes + 255) & ~(size_t)255; return p;
  };
  const size_t BPSLICE = (size_t)(KPAD / 32) * NTPAD * 32 * 16;  // halves per slice
  _Float16* f16A = (_Float16*)alloc((size_t)(NN + 16) * KPAD * 2);
  _Float16* f16B = (_Float16*)alloc(3 * BPSLICE * 2);
  float*    hbuf = (float*)alloc((size_t)NN * LDH * 4);
  float*    el   = (float*)alloc((size_t)NN * HH * 4);
  float*    er   = (float*)alloc((size_t)NN * HH * 4);
  unsigned* mEnc = (unsigned*)alloc((size_t)NN * HH * 4);
  float*    z    = (float*)alloc((size_t)NN * HH * 4);
  float*    exB  = (float*)alloc((size_t)ET * 4);
  float*    fcur = (float*)alloc((size_t)NN * HD * 4);
  float*    f5   = (float*)alloc((size_t)NN * HD * 4);
  float*    agg  = (float*)alloc((size_t)NN * HD * 4);
  float*    p1   = (float*)alloc((size_t)P1R * HD * 4);
  float*    ssum = (float*)alloc((size_t)NB * HD * 4);

  k_zero<<<g1((long)NN * HD), 256, 0, stream>>>(f5, (long)NN * HD);

  // ---- 4 GAT layers (layer 3 weights reused by layer 4) ----
  for (int ll = 0; ll < 4; ++ll) {
    const int Kin  = (ll == 0) ? INF_ : HD;
    const int Kpad = (ll == 0) ? INF_ : KPAD;   // 128 already mult of 32
    const int kT   = Kpad / 32;
    const float* xin = (ll == 0) ? x : fcur;

    long apack = (long)(NN + 16) * Kpad;
    k_pack_a<<<g1(apack), 256, 0, stream>>>(xin, f16A, NN, Kin, Kpad, apack);
    k_pack_w_frag<<<g1((long)kT * NTPAD * 512), 256, 0, stream>>>(
        Wl[ll], f16B, Kin, HD, HD, 1, 0, kT);
    gemm(f16A, Kpad, f16B, hbuf, LDH, NN, NTIL, Kpad, 0, stream);

    k_scores<<<g1((long)NN * HH), 256, 0, stream>>>(hbuf, all[ll], arl[ll], el, er, mEnc, z);
    k_edge_max<<<g1(ET), 256, 0, stream>>>(src, dst, E, el, er, mEnc, ET);
    k_edge_expsum<<<g1(ET), 256, 0, stream>>>(src, dst, E, el, er, mEnc, exB, z, ET);
    k_zero<<<g1((long)NN * HD), 256, 0, stream>>>(agg, (long)NN * HD);
    k_edge_agg<<<g1(ET), 256, 0, stream>>>(src, dst, E, exB, z, hbuf, agg, ET);
    k_bias_relu_acc<<<g1((long)NN * HD), 256, 0, stream>>>(agg, bl[ll], fcur, f5);
  }

  // ---- Conv1d(200,200,3) as 3 row-shifted accumulated GEMMs ----
  const int kT7 = KPAD / 32;
  long apack = (long)(NN + 16) * KPAD;
  k_pack_a<<<g1(apack), 256, 0, stream>>>(f5, f16A, NN, HD, KPAD, apack);
  for (int k = 0; k < 3; ++k)
    k_pack_w_frag<<<g1((long)kT7 * NTPAD * 512), 256, 0, stream>>>(
        cw1, f16B + (size_t)k * BPSLICE, HD, HD, 3, 600, k, kT7);
  for (int k = 0; k < 3; ++k)
    gemm(f16A + (size_t)k * KPAD, KPAD, f16B + (size_t)k * BPSLICE,
         hbuf, LDH, NN, NTIL, KPAD, k > 0, stream);

  k_conv_pool1<<<g1((long)P1R * HD), 256, 0, stream>>>(hbuf, cb1, p1);

  // ---- Conv1d(200,200,1) as one GEMM ----
  long apack2 = (long)P1R * KPAD;
  k_pack_a<<<g1(apack2), 256, 0, stream>>>(p1, f16A, P1R, HD, KPAD, apack2);
  k_pack_w_frag<<<g1((long)kT7 * NTPAD * 512), 256, 0, stream>>>(
      cw2, f16B, HD, HD, 1, HD, 0, kT7);
  gemm(f16A, KPAD, f16B, hbuf, LDH, P1R, NTIL, KPAD, 0, stream);

  k_pool2_sum<<<g1(NB * HD), 256, 0, stream>>>(hbuf, cb2, ssum);
  k_classifier<<<g1(NB, 64), 64, 0, stream>>>(ssum, lw, lb, out);
}
